// H2OLlamaAttention_73469710566028
// MI455X (gfx1250) — matrix operand order repaired and checked by reading, not compiled, not run on hardware
//
#include <hip/hip_runtime.h>
#include <hip/hip_bf16.h>

// ---------------------------------------------------------------------------
// MI455X (gfx1250) Llama GQA attention layer.
//  - All matmuls via v_wmma_f32_16x16x32_bf16 (fp32 accum). Compute-bound:
//    ~417 GFLOP vs ~300 MB of traffic (13 us @ 23.3 TB/s); the bf16 WMMA
//    shape moves 8x the K per issue vs the only-f32 WMMA (16x16x4).
//  - GEMM tiles staged by the Tensor Data Mover (tensor_load_to_lds) with
//    double-buffered LDS: DMA of slab i+1 overlaps WMMA on slab i, fenced by
//    s_wait_tensorcnt + split barrier. Whole bf16 working set (~117 MB) fits
//    in the 192 MB L2, so TDM streams from L2, not HBM.
// ---------------------------------------------------------------------------

#define HIDDEN 4096
#define NH     32
#define NKV    8
#define HD     128
#define BATCH  2
#define SEQ    2048

typedef __attribute__((ext_vector_type(16))) __bf16 bf16x16;
typedef __attribute__((ext_vector_type(8)))  float  f32x8;
typedef __attribute__((ext_vector_type(4)))  unsigned int u32x4;
typedef __attribute__((ext_vector_type(8)))  int    i32x8;
typedef __attribute__((ext_vector_type(4)))  int    i32x4;

__device__ __forceinline__ unsigned short f32_to_bf16(float f) {
  union { float f; unsigned u; } v; v.f = f;
  unsigned u = v.u;
  unsigned r = u + 0x7fffu + ((u >> 16) & 1u);   // round-to-nearest-even
  return (unsigned short)(r >> 16);
}
__device__ __forceinline__ float bf16_to_f32(unsigned short h) {
  union { unsigned u; float f; } v; v.u = ((unsigned)h) << 16; return v.f;
}
__device__ __forceinline__ bf16x16 frag_from(uint4 lo, uint4 hi) {
  union { uint4 u[2]; bf16x16 v; } z; z.u[0] = lo; z.u[1] = hi; return z.v;
}

// ---------------------------------------------------------------------------
// Tensor Data Mover: DMA a 2-D tile (tile_w x tile_h elements of bf16, row
// stride = stride_elems) from global memory into contiguous LDS.
// D# packing per CDNA5 ISA 8.3/8.4:
//   group0: [1:0] count=1 | [63:32] lds_addr | [120:64] global_addr |
//           [127:126] type=2
//   group1: [17:16] data_size=1 (2 B) | [79:48] tensor_dim0 |
//           [111:80] tensor_dim1 | [127:112] tile_dim0 | [143:128] tile_dim1 |
//           [207:160] tensor_dim0_stride
// Tiles are exact (no OOB), so tensor dims == tile dims. Groups 2/3 zero (2D).
// Completion is tracked with TENSORcnt (s_wait_tensorcnt).
// ---------------------------------------------------------------------------
__device__ __forceinline__ void tdm_load_2d(const void* gsrc, void* ldst,
                                            unsigned tile_w, unsigned tile_h,
                                            unsigned stride_elems) {
  unsigned long long ga = (unsigned long long)(uintptr_t)gsrc;
  unsigned lds = (unsigned)(uintptr_t)ldst;    // low 32 bits = LDS offset
  u32x4 g0;
  g0.x = 1u;                                   // count=1, user descriptor
  g0.y = lds;
  g0.z = (unsigned)ga;
  g0.w = (unsigned)((ga >> 32) & 0x01ffffffu) | (2u << 30);  // type=2 (image)
  i32x8 g1;
  g1[0] = (int)(1u << 16);                     // data_size=1 -> 2 bytes
  g1[1] = (int)((tile_w & 0xffffu) << 16);     // tensor_dim0[15:0]
  g1[2] = (int)((tile_w >> 16) | ((tile_h & 0xffffu) << 16)); // dim0 hi, dim1 lo
  g1[3] = (int)((tile_h >> 16) | ((tile_w & 0xffffu) << 16)); // dim1 hi, tile_dim0
  g1[4] = (int)(tile_h & 0xffffu);             // tile_dim1 (tile_dim2=0)
  g1[5] = (int)stride_elems;                   // tensor_dim0_stride lo32
  g1[6] = 0;                                   // stride hi16 | dim1_stride lo16
  g1[7] = 0;
  i32x4 gz = {0, 0, 0, 0};
#if defined(__clang_major__) && (__clang_major__ >= 23)
  i32x8 gz8 = {0, 0, 0, 0, 0, 0, 0, 0};
  __builtin_amdgcn_tensor_load_to_lds(g0, g1, gz, gz, gz8, 0);
#else
  __builtin_amdgcn_tensor_load_to_lds(g0, g1, gz, gz, 0);
#endif
}

// ---------------------------------------------------------------------------
// fp32 -> bf16 elementwise convert (grid-stride)
// ---------------------------------------------------------------------------
__global__ void cvt_f32_bf16_kernel(const float* __restrict__ src,
                                    unsigned short* __restrict__ dst, long n) {
  long i = (long)blockIdx.x * blockDim.x + threadIdx.x;
  long stride = (long)gridDim.x * blockDim.x;
  for (; i < n; i += stride) dst[i] = f32_to_bf16(src[i]);
}

// ---------------------------------------------------------------------------
// In-place RoPE on bf16 tensor laid out [B, S, H, 128].
// Pair (j, j+64): out[j] = x[j]*c - x[j+64]*s ; out[j+64] = x[j+64]*c + x[j]*s
// ---------------------------------------------------------------------------
__global__ void rope_kernel(unsigned short* __restrict__ x,
                            const int* __restrict__ pos,
                            int S, int H, long total) {
  long i = (long)blockIdx.x * blockDim.x + threadIdx.x;
  if (i >= total) return;
  int  j  = (int)(i & 63);
  long t  = i >> 6;                 // (b*S + s)*H + h
  long bs = t / H;
  int  s  = (int)(bs % S);
  long b  = bs / S;
  int  p  = pos[b * S + s];
  float inv = __expf(-(float)(2 * j) * (1.0f / 128.0f) * 9.210340371976184f);
  float ang = (float)p * inv;
  float sn, cs;
  __sincosf(ang, &sn, &cs);
  long base = t * HD;
  float x0 = bf16_to_f32(x[base + j]);
  float x1 = bf16_to_f32(x[base + j + 64]);
  x[base + j]      = f32_to_bf16(x0 * cs - x1 * sn);
  x[base + j + 64] = f32_to_bf16(x1 * cs + x0 * sn);
}

// ---------------------------------------------------------------------------
// bf16 GEMM: C[M,N] = A[M,K] @ B[K,N], row-major, fp32 accumulation.
// Block tile 128x128, 8 waves, wave tile 32x64 (2x4 WMMA accs).
// K staged 64-deep via TDM into double-buffered LDS; DMA of slab i+1 overlaps
// compute on slab i. WMMA fragment addressing per CDNA5 ISA layouts:
//   A (16-bit 16x32): lane=M%16; regs0-3 hold K = 8*(lane/16)+0..7,
//                     regs4-7 hold K = 16+8*(lane/16)+0..7.
//   B (16-bit 32x16): lane%16=N; halves 0-7 K = 8*(lane/16)+0..7, 8-15 K = 16+8*(lane/16)+0..7.
//   C (f32 16x16):    reg r -> M = r + 8*(lane/16); lane%16 -> N.
// ---------------------------------------------------------------------------
template <bool OUT_F32>
__global__ __launch_bounds__(256)
void gemm_bf16_wmma(const unsigned short* __restrict__ A,
                    const unsigned short* __restrict__ Bm,
                    void* __restrict__ C, int M, int N, int K) {
  __shared__ __align__(16) unsigned short As[2][128][64];   // 32 KB
  __shared__ __align__(16) unsigned short Bs[2][64][128];   // 32 KB

  const int tid  = threadIdx.x;
  const int lane = tid & 31;
  const int wave = tid >> 5;
  const int wm   = wave & 3;              // 4 waves along M
  const int wn   = wave >> 2;             // 2 waves along N
  const int bm   = (int)blockIdx.y * 128;
  const int bn   = (int)blockIdx.x * 128;
  const int hl   = lane >> 4;             // lane-half
  const int l16  = lane & 15;
  const int kb   = hl * 8;

  f32x8 acc[2][4];
#pragma unroll
  for (int i = 0; i < 2; i++)
#pragma unroll
    for (int j = 0; j < 4; j++) acc[i][j] = {};

  const int nk = K / 64;
  if (wave == 0) {   // prologue DMA: slab 0 (EXEC-independent, one op/wave)
    tdm_load_2d(&A[(size_t)bm * K], &As[0][0][0], 64, 128, (unsigned)K);
    tdm_load_2d(&Bm[bn], &Bs[0][0][0], 128, 64, (unsigned)N);
  }

  for (int i = 0; i < nk; i++) {
    const int cur = i & 1;
    if (wave == 0) __builtin_amdgcn_s_wait_tensorcnt(0);  // slab i landed
    __syncthreads();                                      // publish to block
    if (wave == 0 && (i + 1) < nk) {   // DMA slab i+1 while computing slab i
      const int k0 = (i + 1) * 64;
      tdm_load_2d(&A[(size_t)bm * K + k0], &As[cur ^ 1][0][0], 64, 128,
                  (unsigned)K);
      tdm_load_2d(&Bm[(size_t)k0 * N + bn], &Bs[cur ^ 1][0][0], 128, 64,
                  (unsigned)N);
    }

#pragma unroll
    for (int ks = 0; ks < 2; ks++) {
      bf16x16 afr[2];
#pragma unroll
      for (int mt = 0; mt < 2; mt++) {
        int row = wm * 32 + mt * 16 + l16;
        int kc  = ks * 32 + kb;
        afr[mt] = frag_from(*(const uint4*)&As[cur][row][kc],
                            *(const uint4*)&As[cur][row][kc + 16]);
      }
#pragma unroll
      for (int nt = 0; nt < 4; nt++) {
        int col = wn * 64 + nt * 16;
        int kr  = ks * 32 + kb;
        bf16x16 bfr;
#pragma unroll
        for (int s = 0; s < 16; s++)
          bfr[s] = __builtin_bit_cast(__bf16, Bs[cur][kr + (s & 7) + ((s >> 3) << 4)][col + l16]);
#pragma unroll
        for (int mt = 0; mt < 2; mt++) {
          acc[mt][nt] = __builtin_amdgcn_wmma_f32_16x16x32_bf16(
              false, afr[mt], false, bfr, (short)0, acc[mt][nt], false, false);
        }
      }
    }
    __syncthreads();   // all reads of slab i done before its buffer is reused
  }

#pragma unroll
  for (int mt = 0; mt < 2; mt++)
#pragma unroll
    for (int nt = 0; nt < 4; nt++) {
      int col = bn + wn * 64 + nt * 16 + l16;
#pragma unroll
      for (int r = 0; r < 8; r++) {
        int row = bm + wm * 32 + mt * 16 + r + hl * 8;
        float v = acc[mt][nt][r];
        if (OUT_F32)
          ((float*)C)[(size_t)row * N + col] = v;
        else
          ((unsigned short*)C)[(size_t)row * N + col] = f32_to_bf16(v);
      }
    }
}

// ---------------------------------------------------------------------------
// Flash-attention (causal, GQA). Block = (b, h, 128 q-rows), 8 waves,
// wave = 16 q-rows. Q tile DMA'd by TDM then lifted to A-fragments; V tile
// DMA'd by TDM (natural layout = PV B-frag contiguity); K staged transposed
// by the waves (TDM cannot transpose; score B-frags need K^T contiguity).
// P re-shaped C-layout -> A-layout through a per-wave LDS tile.
// ---------------------------------------------------------------------------
__global__ __launch_bounds__(256)
void attn_kernel(const unsigned short* __restrict__ Q,   // [B,S,NH,HD]
                 const unsigned short* __restrict__ Kg,  // [B,S,NKV,HD]
                 const unsigned short* __restrict__ Vg,  // [B,S,NKV,HD]
                 unsigned short* __restrict__ O,         // [B,S,NH,HD]
                 int S) {
  __shared__ __align__(16) unsigned short Qs[128][128];   // 32 KB
  __shared__ __align__(16) unsigned short Kt[128][64];    // [dim][key] 16 KB
  __shared__ __align__(16) unsigned short Vs[64][128];    // [key][dim] 16 KB
  __shared__ __align__(16) unsigned short Ps[8][16][64];  // per-wave P 16 KB

  const int tid  = threadIdx.x;
  const int lane = tid & 31;
  const int wave = tid >> 5;
  const int hl   = lane >> 4;
  const int l16  = lane & 15;
  const int kb   = hl * 8;

  const int bh  = (int)blockIdx.y;
  const int b   = bh / NH, h = bh % NH;
  const int kvh = h / (NH / NKV);
  const int q0  = (int)blockIdx.x * 128;
  const float scale = 0.08838834764831845f;   // 1/sqrt(128)

  // Q tile via TDM: 128 rows x 128 dims, row stride NH*HD
  if (wave == 0) {
    tdm_load_2d(&Q[(((size_t)b * S + q0) * NH + h) * HD], &Qs[0][0],
                HD, 128, (unsigned)(NH * HD));
    __builtin_amdgcn_s_wait_tensorcnt(0);
  }
  __syncthreads();

  bf16x16 qfr[4];
  {
    int row = wave * 16 + l16;
#pragma unroll
    for (int d = 0; d < 4; d++) {
      int kc = d * 32 + kb;
      qfr[d] = frag_from(*(const uint4*)&Qs[row][kc],
                         *(const uint4*)&Qs[row][kc + 16]);
    }
  }

  f32x8 o_acc[8];
#pragma unroll
  for (int i = 0; i < 8; i++) o_acc[i] = {};
  float m_st[8], l_st[8];
#pragma unroll
  for (int r = 0; r < 8; r++) { m_st[r] = -3.0e38f; l_st[r] = 0.0f; }

  const int n_tiles = (q0 + 128) / 64;    // causal: key0 <= q0+127

  for (int t = 0; t < n_tiles; t++) {
    const int key0 = t * 64;
    __syncthreads();
    // V tile via TDM: 64 keys x 128 dims, row stride NKV*HD
    if (wave == 0)
      tdm_load_2d(&Vg[(((size_t)b * S + key0) * NKV + kvh) * HD], &Vs[0][0],
                  HD, 64, (unsigned)(NKV * HD));
    // K tile (64 keys x 128 dims) -> transposed Kt[dim][key] by the waves
#pragma unroll
    for (int it = 0; it < 4; it++) {
      int c = tid + it * 256;
      int key = c >> 4, d0 = (c & 15) * 8;
      uint4 u = *(const uint4*)
          &Kg[(((size_t)b * S + key0 + key) * NKV + kvh) * HD + d0];
      const unsigned short* us = (const unsigned short*)&u;
#pragma unroll
      for (int i = 0; i < 8; i++) Kt[d0 + i][key] = us[i];
    }
    if (wave == 0) __builtin_amdgcn_s_wait_tensorcnt(0);
    __syncthreads();

    // scores: 16 q-rows x 64 keys = 4 WMMA tiles, K-dim = 128 = 4 steps
    f32x8 sc[4];
#pragma unroll
    for (int nt = 0; nt < 4; nt++) {
      f32x8 a = {};
#pragma unroll
      for (int d = 0; d < 4; d++) {
        int kr = d * 32 + kb;
        bf16x16 bfr;
#pragma unroll
        for (int s = 0; s < 16; s++)
          bfr[s] = __builtin_bit_cast(__bf16, Kt[kr + (s & 7) + ((s >> 3) << 4)][nt * 16 + l16]);
        a = __builtin_amdgcn_wmma_f32_16x16x32_bf16(
            false, qfr[d], false, bfr, (short)0, a, false, false);
      }
      sc[nt] = a;
    }

    // scale + causal mask
    const int row_base = q0 + wave * 16 + hl * 8;
#pragma unroll
    for (int nt = 0; nt < 4; nt++) {
      int col = key0 + nt * 16 + l16;
#pragma unroll
      for (int r = 0; r < 8; r++) {
        float v = sc[nt][r] * scale;
        if (col > row_base + r) v = -3.0e38f;
        sc[nt][r] = v;
      }
    }

    // online softmax per row (row = 16 lanes of one half-wave)
    float alpha[8];
#pragma unroll
    for (int r = 0; r < 8; r++) {
      float mx = fmaxf(fmaxf(sc[0][r], sc[1][r]), fmaxf(sc[2][r], sc[3][r]));
#pragma unroll
      for (int m = 8; m >= 1; m >>= 1) mx = fmaxf(mx, __shfl_xor(mx, m, 32));
      float mnew = fmaxf(m_st[r], mx);
      float al   = __expf(m_st[r] - mnew);
      float sum  = 0.0f;
#pragma unroll
      for (int nt = 0; nt < 4; nt++) {
        float p = __expf(sc[nt][r] - mnew);
        sc[nt][r] = p;
        sum += p;
      }
#pragma unroll
      for (int m = 8; m >= 1; m >>= 1) sum += __shfl_xor(sum, m, 32);
      m_st[r] = mnew;
      l_st[r] = l_st[r] * al + sum;
      alpha[r] = al;
    }
#pragma unroll
    for (int i = 0; i < 8; i++)
#pragma unroll
      for (int r = 0; r < 8; r++) o_acc[i][r] *= alpha[r];

    // P: C-layout f32 -> bf16 A-layout via per-wave LDS tile
#pragma unroll
    for (int nt = 0; nt < 4; nt++)
#pragma unroll
      for (int r = 0; r < 8; r++)
        Ps[wave][r + hl * 8][nt * 16 + l16] = f32_to_bf16(sc[nt][r]);
    __syncthreads();

    // O += P @ V  (two 32-key steps, 8 dim tiles)
#pragma unroll
    for (int kt = 0; kt < 2; kt++) {
      int kc = kt * 32 + kb;
      bf16x16 pfr = frag_from(*(const uint4*)&Ps[wave][l16][kc],
                              *(const uint4*)&Ps[wave][l16][kc + 16]);
#pragma unroll
      for (int nt2 = 0; nt2 < 8; nt2++) {
        int kr = kt * 32 + kb;
        bf16x16 vfr;
#pragma unroll
        for (int s = 0; s < 16; s++)
          vfr[s] = __builtin_bit_cast(__bf16, Vs[kr + (s & 7) + ((s >> 3) << 4)][nt2 * 16 + l16]);
        o_acc[nt2] = __builtin_amdgcn_wmma_f32_16x16x32_bf16(
            false, pfr, false, vfr, (short)0, o_acc[nt2], false, false);
      }
    }
  }

  // finalize: divide by row sums, write bf16 [b,s,h,d]
#pragma unroll
  for (int nt2 = 0; nt2 < 8; nt2++)
#pragma unroll
    for (int r = 0; r < 8; r++) {
      int row = q0 + wave * 16 + r + hl * 8;
      float v = o_acc[nt2][r] / l_st[r];
      O[(((size_t)b * S + row) * NH + h) * HD + nt2 * 16 + l16] =
          f32_to_bf16(v);
    }
}

// ---------------------------------------------------------------------------
// Host-side orchestration. Workspace layout (~168 MB, all bf16):
//   Xb[4096x4096] | Wqb[4096x4096] | Wkb[4096x1024] | Wvb[4096x1024] |
//   Wob[4096x4096] | Qb[4096x4096] | Kb[4096x1024] | Vb[4096x1024]
// Attention output aliases Xb (X is dead after QKV projections).
// ---------------------------------------------------------------------------
extern "C" void kernel_launch(void* const* d_in, const int* in_sizes, int n_in,
                              void* d_out, int out_size, void* d_ws,
                              size_t ws_size, hipStream_t stream) {
  (void)in_sizes; (void)n_in; (void)out_size; (void)ws_size;
  const float* hs  = (const float*)d_in[0];
  const int*   pos = (const int*)d_in[1];
  const float* Wq  = (const float*)d_in[2];
  const float* Wk  = (const float*)d_in[3];
  const float* Wv  = (const float*)d_in[4];
  const float* Wo  = (const float*)d_in[5];

  const int  M  = BATCH * SEQ;              // 4096
  const long nH = (long)M * HIDDEN;         // 16.7M
  const long nW = (long)HIDDEN * HIDDEN;
  const long nK = (long)HIDDEN * (NKV * HD);

  char* p = (char*)d_ws;
  unsigned short* Xb  = (unsigned short*)p; p += nH * 2;
  unsigned short* Wqb = (unsigned short*)p; p += nW * 2;
  unsigned short* Wkb = (unsigned short*)p; p += nK * 2;
  unsigned short* Wvb = (unsigned short*)p; p += nK * 2;
  unsigned short* Wob = (unsigned short*)p; p += nW * 2;
  unsigned short* Qb  = (unsigned short*)p; p += nH * 2;
  unsigned short* Kb  = (unsigned short*)p; p += (long)M * NKV * HD * 2;
  unsigned short* Vb  = (unsigned short*)p; p += (long)M * NKV * HD * 2;
  unsigned short* Ob  = Xb;                 // reuse X after QKV GEMMs

  // 1) fp32 -> bf16
  cvt_f32_bf16_kernel<<<8192, 256, 0, stream>>>(hs, Xb, nH);
  cvt_f32_bf16_kernel<<<8192, 256, 0, stream>>>(Wq, Wqb, nW);
  cvt_f32_bf16_kernel<<<4096, 256, 0, stream>>>(Wk, Wkb, nK);
  cvt_f32_bf16_kernel<<<4096, 256, 0, stream>>>(Wv, Wvb, nK);
  cvt_f32_bf16_kernel<<<8192, 256, 0, stream>>>(Wo, Wob, nW);

  // 2) QKV projections (bf16 out)
  gemm_bf16_wmma<false><<<dim3(HIDDEN / 128, M / 128), 256, 0, stream>>>(
      Xb, Wqb, Qb, M, HIDDEN, HIDDEN);
  gemm_bf16_wmma<false><<<dim3((NKV * HD) / 128, M / 128), 256, 0, stream>>>(
      Xb, Wkb, Kb, M, NKV * HD, HIDDEN);
  gemm_bf16_wmma<false><<<dim3((NKV * HD) / 128, M / 128), 256, 0, stream>>>(
      Xb, Wvb, Vb, M, NKV * HD, HIDDEN);

  // 3) RoPE on Q and K (in place)
  long nQr = (long)M * NH * 64;
  long nKr = (long)M * NKV * 64;
  rope_kernel<<<(unsigned)((nQr + 255) / 256), 256, 0, stream>>>(
      Qb, pos, SEQ, NH, nQr);
  rope_kernel<<<(unsigned)((nKr + 255) / 256), 256, 0, stream>>>(
      Kb, pos, SEQ, NKV, nKr);

  // 4) causal flash attention (bf16 out, aliases Xb)
  attn_kernel<<<dim3(SEQ / 128, BATCH * NH), 256, 0, stream>>>(
      Qb, Kb, Vb, Ob, SEQ);

  // 5) output projection -> fp32 d_out
  gemm_bf16_wmma<true><<<dim3(HIDDEN / 128, M / 128), 256, 0, stream>>>(
      Ob, Wob, d_out, M, HIDDEN, HIDDEN);
}
